// AnomalyDetector_12790412607793
// MI455X (gfx1250) — matrix operand order repaired and checked by reading, not compiled, not run on hardware
//
#include <hip/hip_runtime.h>
#include <hip/hip_bf16.h>
#include <math.h>

// ---------------- problem constants ----------------
static constexpr int       B_   = 16;
static constexpr int       S_   = 4096;
static constexpr int       IN_  = 64;
static constexpr int       D_   = 256;
static constexpr int       N_   = 16;
static constexpr int       K_   = 32;
static constexpr int       F_   = 2049;   // rfft bins
static constexpr int       FP_  = 2112;   // padded to 33*64 for 64-row wave tiles
static constexpr int       NC_  = 288;    // [Wd|WB|WC] concat cols (9*32)
static constexpr long long BS_  = (long long)B_ * S_;       // 65536
static constexpr long long BSD_ = BS_ * D_;                 // 16.8M

typedef __attribute__((ext_vector_type(16))) __bf16 v16bf;
typedef __attribute__((ext_vector_type(8)))  __bf16 v8bf;
typedef __attribute__((ext_vector_type(8)))  float  v8f;

__device__ __forceinline__ float silu_f(float x) { return x / (1.f + __expf(-x)); }
__device__ __forceinline__ float softplus_f(float x) {
  if (x > 20.f) return x;
  if (x < -20.f) return __expf(x);
  return log1pf(__expf(x));
}

// ---------------- elementwise f32 -> bf16 ----------------
__global__ void k_cvt_bf16(__bf16* __restrict__ dst, const float* __restrict__ src, long long n) {
  long long i = (long long)blockIdx.x * blockDim.x + threadIdx.x;
  if (i < n) dst[i] = (__bf16)src[i];
}

// ---------------- pack B matrix [K x N] f32 into WMMA bf16 fragment order --------
// Fragment layout (16-bit B, 32x16 tile, wave32): lane l holds column N = l&15,
// elements e<8 are K = (l>>4)*8 + e, e>=8 are K = 16 + (l>>4)*8 + (e-8).  Packed index:
//   (((nt16*KT + kt)*32 + lane)*16 + e)
__global__ void k_pack_b(__bf16* __restrict__ dst, const float* __restrict__ src, int Kd, int Nd) {
  long long i = (long long)blockIdx.x * blockDim.x + threadIdx.x;
  long long tot = (long long)Kd * Nd;
  if (i >= tot) return;
  int e    = (int)(i & 15);
  int lane = (int)((i >> 4) & 31);
  long long rest = i >> 9;
  int KT = Kd >> 5;
  int kt = (int)(rest % KT);
  int nt = (int)(rest / KT);
  int krow = kt * 32 + (lane >> 4) * 8 + (e < 8 ? e : 8 + e);
  int col  = nt * 16 + (lane & 15);
  dst[i] = (__bf16)src[(long long)krow * Nd + col];
}

// ---------------- build [Wd|WB|WC] concat (f32) + concat bias ----------------
__global__ void k_build_wcat(float* __restrict__ w, float* __restrict__ bias,
                             const float* Wd, const float* bd,
                             const float* WB, const float* bB,
                             const float* WC, const float* bC) {
  int i = blockIdx.x * blockDim.x + threadIdx.x;
  if (i < NC_) bias[i] = (i < D_) ? bd[i] : (i < D_ + N_) ? bB[i - D_] : bC[i - D_ - N_];
  if (i >= D_ * NC_) return;
  int r = i / NC_, c = i % NC_;
  float v = (c < D_) ? Wd[r * D_ + c]
          : (c < D_ + N_) ? WB[r * N_ + (c - D_)]
          : WC[r * N_ + (c - D_ - N_)];
  w[i] = v;
}

// ---------------- core WMMA GEMM: C[M,N] = act(A[M,K] @ Bpacked + bias) ----------------
// A: bf16 row-major (M % 64 == 0).  Bp: bf16 packed (see k_pack_b).
// One wave computes a 64x32 tile: 4 m-subtiles x 2 n-subtiles = 8 WMMAs per K-step.
// ACT: 0 = none, 1 = softplus on columns < 256 (delta), identity beyond (Bm/Cm).
template <int ACT, bool HAS_BIAS>
__global__ void k_gemm_wmma_bf16(float* __restrict__ C, const __bf16* __restrict__ A,
                                 const __bf16* __restrict__ Bp, const float* __restrict__ bias,
                                 int M, int Nd, int Kd) {
  const int lane = threadIdx.x & 31;
  const int wave = (int)((blockIdx.x * blockDim.x + threadIdx.x) >> 5);
  const int nt32 = Nd >> 5;
  const int mt = wave / nt32;
  const int nt = wave % nt32;
  if (mt * 64 >= M) return;

  const int r  = lane & 15;   // A row within 16-subtile / C column within 16-subtile
  const int hi = lane >> 4;
  const int KT = Kd >> 5;

  v8f acc[4][2] = {};

  const __bf16* a0 = A + (long long)(mt * 64 + r) * Kd + hi * 8;
  const __bf16* b0 = Bp + (long long)(nt * 2 * KT + 0) * 512 + lane * 16;

  for (int kt = 0; kt < KT; ++kt) {
    const int k0 = kt * 32;
    v16bf af[4];
#pragma unroll
    for (int mi = 0; mi < 4; ++mi) {
      // 16-bit A 16x32: lane half 'hi' owns K runs [hi*8, hi*8+7] and [16+hi*8, 23+hi*8]
      const __bf16* ap = a0 + (long long)mi * 16 * Kd + k0;
      __builtin_prefetch(ap + 128, 0, 1);            // global_prefetch_b8 ahead
      v8bf lo  = *(const v8bf*)ap;
      v8bf hi8 = *(const v8bf*)(ap + 16);
#pragma unroll
      for (int e = 0; e < 8; ++e) { af[mi][e] = lo[e]; af[mi][e + 8] = hi8[e]; }
    }
    v16bf bf[2];
#pragma unroll
    for (int ni = 0; ni < 2; ++ni)
      bf[ni] = *(const v16bf*)(b0 + (long long)(ni * KT + kt) * 512); // 32B/lane
#pragma unroll
    for (int mi = 0; mi < 4; ++mi)
#pragma unroll
      for (int ni = 0; ni < 2; ++ni)
        acc[mi][ni] = __builtin_amdgcn_wmma_f32_16x16x32_bf16(
            false, af[mi], false, bf[ni], (short)0, acc[mi][ni], false, false);
  }

  // hoisted per-lane bias (one column per ni-subtile)
  float bcol[2];
#pragma unroll
  for (int ni = 0; ni < 2; ++ni)
    bcol[ni] = HAS_BIAS ? bias[nt * 32 + ni * 16 + r] : 0.f;

  // C/D layout: VGPR v -> M = 8*(lane>>4) + v ; N = lane&15
#pragma unroll
  for (int mi = 0; mi < 4; ++mi)
#pragma unroll
    for (int ni = 0; ni < 2; ++ni) {
      const int col = nt * 32 + ni * 16 + r;
      float* cp = C + (long long)(mt * 64 + mi * 16 + hi * 8) * Nd + col;
#pragma unroll
      for (int v = 0; v < 8; ++v) {
        float val = acc[mi][ni][v] + bcol[ni];
        if (ACT == 1 && col < D_) val = softplus_f(val);
        cp[(long long)v * Nd] = val;
      }
    }
}

// ---------------- LayerNorm over D=256, one wave per token ----------------
__global__ void k_layernorm(float* __restrict__ xn, __bf16* __restrict__ xn_bf,
                            const float* __restrict__ xp,
                            const float* __restrict__ g, const float* __restrict__ b) {
  int tok  = blockIdx.x * (blockDim.x >> 5) + (threadIdx.x >> 5);
  int lane = threadIdx.x & 31;
  if (tok >= (int)BS_) return;
  const float* row = xp + (long long)tok * D_;
  float s = 0.f, q = 0.f;
#pragma unroll
  for (int j = 0; j < 8; ++j) {
    float v = row[lane * 8 + j];
    s += v; q += v * v;
  }
#pragma unroll
  for (int m = 16; m >= 1; m >>= 1) {
    s += __shfl_xor(s, m, 32);
    q += __shfl_xor(q, m, 32);
  }
  float mean = s * (1.f / D_);
  float var  = q * (1.f / D_) - mean * mean;
  float rs   = rsqrtf(var + 1e-5f);
#pragma unroll
  for (int j = 0; j < 8; ++j) {
    int d = lane * 8 + j;
    float o = (row[d] - mean) * rs * g[d] + b[d];
    xn[(long long)tok * D_ + d]    = o;
    xn_bf[(long long)tok * D_ + d] = (__bf16)o;
  }
}

// ---------------- depthwise conv3 (pad 1) + SiLU ----------------
__global__ void k_dwconv_silu(float* __restrict__ u, __bf16* __restrict__ u_bf,
                              const float* __restrict__ xn,
                              const float* __restrict__ w, const float* __restrict__ cb) {
  long long i = (long long)blockIdx.x * blockDim.x + threadIdx.x;
  if (i >= BSD_) return;
  int d = (int)(i % D_);
  long long bt = i / D_;
  int t = (int)(bt % S_);
  float acc = cb[d];
  if (t > 0)        acc += w[d * 3 + 0] * xn[i - D_];
  acc += w[d * 3 + 1] * xn[i];
  if (t < S_ - 1)   acc += w[d * 3 + 2] * xn[i + D_];
  float r = silu_f(acc);
  u[i] = r; u_bf[i] = (__bf16)r;
}

// ---------------- DFT twiddle matrices (bf16), Tre=cos(-2pi f t/S), Tim=sin(-...) ----
__global__ void k_twiddle(__bf16* __restrict__ tre, __bf16* __restrict__ tim) {
  long long i = (long long)blockIdx.x * blockDim.x + threadIdx.x;
  if (i >= (long long)FP_ * S_) return;
  int t = (int)(i % S_);
  int f = (int)(i / S_);
  int ph = (int)(((long long)f * t) & (S_ - 1));
  float ang = -6.283185307179586f * (float)ph / (float)S_;
  float sn, cs; __sincosf(ang, &sn, &cs);
  tre[i] = (__bf16)cs;
  tim[i] = (__bf16)sn;
}

// ---------------- selective-state scan: one thread per (b,d), N=16 state in regs ----
__global__ void k_scan(float* __restrict__ y, const float* __restrict__ u,
                       const float* __restrict__ dcat, const float* __restrict__ Amat) {
  int b = blockIdx.x;
  int d = threadIdx.x;
  float h[N_], a[N_];
#pragma unroll
  for (int j = 0; j < N_; ++j) { h[j] = 0.f; a[j] = Amat[d * N_ + j]; }
  const float* ub = u    + (long long)b * S_ * D_;
  const float* db = dcat + (long long)b * S_ * NC_;
  float*       yb = y    + (long long)b * S_ * D_;
  for (int t = 0; t < S_; ++t) {
    const float* row = db + (long long)t * NC_;
    float ut = ub[(long long)t * D_ + d];
    float dt = row[d];
    float yv = 0.f;
#pragma unroll
    for (int j = 0; j < N_; ++j) {
      float Bv = row[D_ + j];
      float Cv = row[D_ + N_ + j];
      h[j] = fmaf(dt * a[j], h[j], h[j]) + dt * Bv * ut;
      yv  = fmaf(Cv, h[j], yv);
    }
    yb[(long long)t * D_ + d] = yv;
  }
}

// ---------------- x_t = xp + x_ssm * silu(gpre)   (in place on y) ----------------
__global__ void k_fuse_xt(float* __restrict__ y, const float* __restrict__ xp,
                          const float* __restrict__ gpre) {
  long long i = (long long)blockIdx.x * blockDim.x + threadIdx.x;
  if (i >= BSD_) return;
  y[i] = xp[i] + y[i] * silu_f(gpre[i]);
}

// ---------------- top-K |X| per (b,d): one wave per column, 65 slots/lane ----------
__global__ void k_topk(int* __restrict__ fidx, float* __restrict__ fre, float* __restrict__ fim,
                       const float* __restrict__ Xre, const float* __restrict__ Xim,
                       const float* __restrict__ filt_re, const float* __restrict__ filt_im) {
  int w    = (int)((blockIdx.x * blockDim.x + threadIdx.x) >> 5);
  int lane = threadIdx.x & 31;
  if (w >= B_ * D_) return;
  int b = w / D_, d = w % D_;
  const float* xr = Xre + (long long)b * FP_ * D_ + d;
  const float* xi = Xim + (long long)b * FP_ * D_ + d;
  float m[65];
#pragma unroll
  for (int i = 0; i < 65; ++i) {
    int f = i * 32 + lane;
    if (f < F_) {
      float re = xr[(long long)f * D_], im = xi[(long long)f * D_];
      m[i] = re * re + im * im;
    } else m[i] = -1.f;
  }
  for (int k = 0; k < K_; ++k) {
    float best = -2.f; int bi = 0x7fffffff;
#pragma unroll
    for (int i = 0; i < 65; ++i) {
      int f = i * 32 + lane;
      if (m[i] > best) { best = m[i]; bi = f; }
    }
#pragma unroll
    for (int mm = 16; mm >= 1; mm >>= 1) {
      float ov = __shfl_xor(best, mm, 32);
      int   of = __shfl_xor(bi,   mm, 32);
      if (ov > best || (ov == best && of < bi)) { best = ov; bi = of; }
    }
    int slot = bi >> 5;
#pragma unroll
    for (int i = 0; i < 65; ++i)
      if (i == slot && lane == (bi & 31)) m[i] = -2.f;   // mask winner
    if (lane == 0) {
      float re = xr[(long long)bi * D_], im = xi[(long long)bi * D_];
      float fr = filt_re[k * D_ + d],    fi = filt_im[k * D_ + d];
      long long o = (long long)w * K_ + k;
      fidx[o] = bi;
      fre[o]  = re * fr - im * fi;      // (re+i*im)*(fr+i*fi)
      fim[o]  = re * fi + im * fr;
    }
  }
}

// ---------------- sparse irfft + fusion + temporal mean -> zp[b,d] ----------------
__global__ void k_spectral_fuse(float* __restrict__ zp, const float* __restrict__ xt,
                                const int* __restrict__ fidx, const float* __restrict__ fre,
                                const float* __restrict__ fim,
                                const float* __restrict__ alpha, const float* __restrict__ beta) {
  __shared__ float tc[S_];          // 4096 cos
  __shared__ float ts[S_];          // 4096 sin
  __shared__ int   sidx[K_];
  __shared__ float sr[K_], si[K_];
  __shared__ float red[256];

  int b = blockIdx.x / D_, d = blockIdx.x % D_;
  int tid = threadIdx.x;

  for (int i = tid; i < S_; i += blockDim.x) {
    float sn, cs; __sincosf(6.283185307179586f * (float)i / (float)S_, &sn, &cs);
    tc[i] = cs; ts[i] = sn;
  }
  if (tid < K_) {
    long long o = (long long)(b * D_ + d) * K_ + tid;
    int f = fidx[o];
    float wgt = ((f == 0) || (f == S_ / 2)) ? 1.f : 2.f;
    wgt *= (1.f / (float)S_);                 // irfft 1/S
    sidx[tid] = f;
    sr[tid] = fre[o] * wgt;
    si[tid] = fim[o] * wgt;
  }
  __syncthreads();

  float av = alpha[d], bv = beta[d];
  float acc = 0.f;
  for (int t = tid; t < S_; t += blockDim.x) {
    float xf = 0.f;
#pragma unroll
    for (int k = 0; k < K_; ++k) {
      int ph = (sidx[k] * t) & (S_ - 1);
      xf += sr[k] * tc[ph] - si[k] * ts[ph];
    }
    acc += av * xt[((long long)b * S_ + t) * D_ + d] + bv * xf;
  }
  red[tid] = acc;
  __syncthreads();
  for (int s = 128; s > 0; s >>= 1) {
    if (tid < s) red[tid] += red[tid + s];
    __syncthreads();
  }
  if (tid == 0) zp[b * D_ + d] = red[0] * (1.f / (float)S_);   // temporal mean
}

// ---------------- classifier: LN -> ReLU(zp@W1+b1) -> @W2+b2 ----------------
__global__ void k_classifier(float* __restrict__ out, const float* __restrict__ zp,
                             const float* __restrict__ g, const float* __restrict__ bb,
                             const float* __restrict__ W1, const float* __restrict__ b1,
                             const float* __restrict__ W2, const float* __restrict__ b2) {
  __shared__ float zn[B_ * D_];
  __shared__ float hb[B_ * (D_ / 2)];
  int tid = threadIdx.x;
  if (tid < B_) {
    const float* row = zp + tid * D_;
    float s = 0.f, q = 0.f;
    for (int j = 0; j < D_; ++j) { float v = row[j]; s += v; q += v * v; }
    float mean = s / D_;
    float var  = q / D_ - mean * mean;
    float rs   = rsqrtf(var + 1e-5f);
    for (int j = 0; j < D_; ++j) zn[tid * D_ + j] = (row[j] - mean) * rs * g[j] + bb[j];
  }
  __syncthreads();
  for (int idx = tid; idx < B_ * (D_ / 2); idx += blockDim.x) {
    int r = idx / (D_ / 2), c = idx % (D_ / 2);
    float acc = b1[c];
    for (int j = 0; j < D_; ++j) acc += zn[r * D_ + j] * W1[j * (D_ / 2) + c];
    hb[idx] = fmaxf(acc, 0.f);
  }
  __syncthreads();
  if (tid < B_ * 2) {
    int r = tid >> 1, o = tid & 1;
    float acc = b2[o];
    for (int j = 0; j < D_ / 2; ++j) acc += hb[r * (D_ / 2) + j] * W2[j * 2 + o];
    out[tid] = acc;
  }
}

// =====================================================================
extern "C" void kernel_launch(void* const* d_in, const int* in_sizes, int n_in,
                              void* d_out, int out_size, void* d_ws, size_t ws_size,
                              hipStream_t stream) {
  const float* x       = (const float*)d_in[0];
  const float* W_in    = (const float*)d_in[1];
  const float* b_in    = (const float*)d_in[2];
  const float* ln1_g   = (const float*)d_in[3];
  const float* ln1_b   = (const float*)d_in[4];
  const float* conv_w  = (const float*)d_in[5];
  const float* conv_b  = (const float*)d_in[6];
  const float* Wd      = (const float*)d_in[7];
  const float* bd      = (const float*)d_in[8];
  const float* WB      = (const float*)d_in[9];
  const float* bB      = (const float*)d_in[10];
  const float* WC      = (const float*)d_in[11];
  const float* bC      = (const float*)d_in[12];
  const float* Amat    = (const float*)d_in[13];
  const float* Wg      = (const float*)d_in[14];
  const float* bg      = (const float*)d_in[15];
  const float* filt_re = (const float*)d_in[16];
  const float* filt_im = (const float*)d_in[17];
  const float* alpha   = (const float*)d_in[18];
  const float* beta    = (const float*)d_in[19];
  const float* lnc_g   = (const float*)d_in[20];
  const float* lnc_b   = (const float*)d_in[21];
  const float* W1      = (const float*)d_in[22];
  const float* b1      = (const float*)d_in[23];
  const float* W2      = (const float*)d_in[24];
  const float* b2      = (const float*)d_in[25];
  (void)in_sizes; (void)n_in; (void)out_size; (void)ws_size;

  char* base = (char*)d_ws;
  size_t off = 0;
  auto alloc = [&](size_t bytes) -> void* {
    void* p = base + off;
    off = (off + bytes + 255) & ~(size_t)255;
    return p;
  };

  // f32 workspace
  float* xp    = (float*)alloc(BSD_ * 4);
  float* xn    = (float*)alloc(BSD_ * 4);
  float* u     = (float*)alloc(BSD_ * 4);
  float* gpre  = (float*)alloc(BSD_ * 4);
  float* y     = (float*)alloc(BSD_ * 4);                 // x_ssm, then x_t
  float* dcat  = (float*)alloc(BS_ * NC_ * 4);            // [delta|Bm|Cm]
  float* Xre   = (float*)alloc((long long)B_ * FP_ * D_ * 4);
  float* Xim   = (float*)alloc((long long)B_ * FP_ * D_ * 4);
  float* wcatf = (float*)alloc((long long)D_ * NC_ * 4);
  float* bcat  = (float*)alloc(NC_ * 4);
  float* zp    = (float*)alloc((long long)B_ * D_ * 4);
  float* fre   = (float*)alloc((long long)B_ * D_ * K_ * 4);
  float* fim   = (float*)alloc((long long)B_ * D_ * K_ * 4);
  int*   fidx  = (int*)  alloc((long long)B_ * D_ * K_ * 4);
  // bf16 workspace
  __bf16* x_bf   = (__bf16*)alloc(BS_ * IN_ * 2);
  __bf16* xn_bf  = (__bf16*)alloc(BSD_ * 2);
  __bf16* u_bf   = (__bf16*)alloc(BSD_ * 2);
  __bf16* Tre    = (__bf16*)alloc((long long)FP_ * S_ * 2);
  __bf16* Tim    = (__bf16*)alloc((long long)FP_ * S_ * 2);
  __bf16* pkWin  = (__bf16*)alloc((long long)IN_ * D_ * 2);
  __bf16* pkWcat = (__bf16*)alloc((long long)D_ * NC_ * 2);
  __bf16* pkWg   = (__bf16*)alloc((long long)D_ * D_ * 2);
  __bf16* pkXp   = (__bf16*)alloc(BSD_ * 2);              // packed xp per batch

  const int TB = 256;
  auto gemmBlocks = [](long long M, int Nd) -> int {
    return (int)((M / 64) * (Nd / 32) / 8);   // 8 waves per block
  };

  // 1) xp = x @ W_in + b_in
  k_cvt_bf16<<<(int)((BS_ * IN_ + TB - 1) / TB), TB, 0, stream>>>(x_bf, x, BS_ * IN_);
  k_pack_b<<<(IN_ * D_ + TB - 1) / TB, TB, 0, stream>>>(pkWin, W_in, IN_, D_);
  k_gemm_wmma_bf16<0, true><<<gemmBlocks(BS_, D_), TB, 0, stream>>>(
      xp, x_bf, pkWin, b_in, (int)BS_, D_, IN_);
  // 2) LN + depthwise conv + SiLU
  k_layernorm<<<(int)BS_ / 8, TB, 0, stream>>>(xn, xn_bf, xp, ln1_g, ln1_b);
  k_dwconv_silu<<<(int)(BSD_ / TB), TB, 0, stream>>>(u, u_bf, xn, conv_w, conv_b);
  // 3) [delta|Bm|Cm] = act(u @ [Wd|WB|WC] + bias)
  k_build_wcat<<<(D_ * NC_ + TB - 1) / TB, TB, 0, stream>>>(wcatf, bcat, Wd, bd, WB, bB, WC, bC);
  k_pack_b<<<(D_ * NC_ + TB - 1) / TB, TB, 0, stream>>>(pkWcat, wcatf, D_, NC_);
  k_gemm_wmma_bf16<1, true><<<gemmBlocks(BS_, NC_), TB, 0, stream>>>(
      dcat, u_bf, pkWcat, bcat, (int)BS_, NC_, D_);
  // 4) gpre = xn @ Wg + bg
  k_pack_b<<<(D_ * D_ + TB - 1) / TB, TB, 0, stream>>>(pkWg, Wg, D_, D_);
  k_gemm_wmma_bf16<0, true><<<gemmBlocks(BS_, D_), TB, 0, stream>>>(
      gpre, xn_bf, pkWg, bg, (int)BS_, D_, D_);
  // 5) sequential SSM scan, then x_t = xp + x_ssm * silu(gpre)
  k_scan<<<B_, D_, 0, stream>>>(y, u, dcat, Amat);
  k_fuse_xt<<<(int)(BSD_ / TB), TB, 0, stream>>>(y, xp, gpre);
  // 6) rfft as DFT GEMM (bf16 twiddles x packed xp), per batch
  k_twiddle<<<(int)(((long long)FP_ * S_ + TB - 1) / TB), TB, 0, stream>>>(Tre, Tim);
  for (int b = 0; b < B_; ++b) {
    k_pack_b<<<(S_ * D_ + TB - 1) / TB, TB, 0, stream>>>(
        pkXp + (long long)b * S_ * D_, xp + (long long)b * S_ * D_, S_, D_);
  }
  const int dftBlocks = gemmBlocks(FP_, D_);   // 33 blocks of 8 waves
  for (int b = 0; b < B_; ++b) {
    k_gemm_wmma_bf16<0, false><<<dftBlocks, TB, 0, stream>>>(
        Xre + (long long)b * FP_ * D_, Tre, pkXp + (long long)b * S_ * D_, nullptr, FP_, D_, S_);
    k_gemm_wmma_bf16<0, false><<<dftBlocks, TB, 0, stream>>>(
        Xim + (long long)b * FP_ * D_, Tim, pkXp + (long long)b * S_ * D_, nullptr, FP_, D_, S_);
  }
  // 7) top-K select + filter, sparse irfft fused with mean pooling
  k_topk<<<(B_ * D_) / 8, TB, 0, stream>>>(fidx, fre, fim, Xre, Xim, filt_re, filt_im);
  k_spectral_fuse<<<B_ * D_, TB, 0, stream>>>(zp, y, fidx, fre, fim, alpha, beta);
  // 8) classifier head
  k_classifier<<<1, TB, 0, stream>>>((float*)d_out, zp, lnc_g, lnc_b, W1, b1, W2, b2);
}
